// Qwen3GroupedAttention_24678882083046
// MI455X (gfx1250) — compile-verified, matrix operand order checked by reading, not run on hardware
//
#include <hip/hip_runtime.h>
#include <hip/hip_bf16.h>
#include <math.h>
#include <stdint.h>

typedef __attribute__((ext_vector_type(16))) __bf16 v16bf;
typedef __attribute__((ext_vector_type(8)))  float  v8f;
typedef __attribute__((ext_vector_type(4)))  unsigned int v4u;
typedef __attribute__((ext_vector_type(8)))  int    v8i_t;
typedef __attribute__((ext_vector_type(4)))  int    v4i_t;

#if defined(__has_builtin)
#  if __has_builtin(__builtin_amdgcn_tensor_load_to_lds)
#    define USE_TDM 1
#  endif
#endif
#ifndef USE_TDM
#  define USE_TDM 0
#endif

// Build a bf16 A/B fragment from an fp32 tile laid out [row][k], k contiguous.
// src must already include the + ((lane>>4)<<3) K offset.
// Element e of the fragment holds K = (e&7) + ((e>>3)<<4) + ((lane>>4)<<3).
__device__ __forceinline__ v16bf make_frag(const float* src) {
  union { v16bf v; __bf16 e[16]; } f;
#pragma unroll
  for (int e = 0; e < 16; e += 2) {
    float2 x = *(const float2*)(src + (e & 7) + ((e >> 3) << 4));
    f.e[e]     = (__bf16)x.x;
    f.e[e + 1] = (__bf16)x.y;
  }
  return f.v;
}

// Per-lane async copy of 16 bytes: global (saddr + voff) -> LDS[lds_addr]. ASYNCcnt-tracked.
__device__ __forceinline__ void async_copy_b128(uint32_t lds_addr, uint32_t voff, uint64_t saddr) {
  asm volatile("global_load_async_to_lds_b128 %0, %1, %2"
               :: "v"(lds_addr), "v"(voff), "s"(saddr) : "memory");
}
__device__ __forceinline__ void wait_asynccnt0() {
  asm volatile("s_wait_asynccnt 0" ::: "memory");
}

#if USE_TDM
// TDM: load a 2D tile (tile_d1 rows x tile_d0 fp32 elements, row stride stride_elems)
// from global into LDS at lds_addr, rows packed contiguously. TENSORcnt-tracked.
// This toolchain's builtin takes 6 args: (g0 u32x4, g1 i32x8, g2 i32x4, g3 i32x4, i32x8, cpol).
__device__ __forceinline__ void tdm_load_2d(uint32_t lds_addr, uint64_t gaddr,
                                            int tile_d1, int tile_d0, int stride_elems) {
  v4u g0;
  g0.x = 1u;                                                 // count=1, user mode
  g0.y = lds_addr;                                           // LDS byte address
  g0.z = (uint32_t)(gaddr & 0xFFFFFFFFu);                    // global_addr[31:0]
  g0.w = (uint32_t)((gaddr >> 32) & 0x01FFFFFFu) | 0x80000000u; // addr[56:32] | type=2
  v8i_t g1;
  g1[0] = (int)(2u << 16);                                   // wg_mask=0, data_size=4B
  g1[1] = (int)((uint32_t)tile_d0 << 16);                    // tensor_dim0[15:0] @ [63:48]
  g1[2] = (int)((uint32_t)tile_d1 << 16);                    // tensor_dim1[15:0] @ [95:80]
  g1[3] = (int)((uint32_t)tile_d0 << 16);                    // tile_dim0 @ [127:112]
  g1[4] = tile_d1;                                           // tile_dim1 @ [143:128]
  g1[5] = stride_elems;                                      // tensor_dim0_stride[31:0]
  g1[6] = 0;
  g1[7] = 0;
  v4i_t z4 = {0, 0, 0, 0};
  v8i_t z8 = {0, 0, 0, 0, 0, 0, 0, 0};
  __builtin_amdgcn_tensor_load_to_lds(g0, g1, z4, z4, z8, 0);
}
#endif

// C = A[MxK] * B[KxN], row-major fp32 in/out, bf16 WMMA compute, f32 accumulate.
// Block tile 128x128, 8 waves in 2(M) x 4(N), wave tile 64x32 (4x2 subtiles).
// A tile staged via async global->LDS; B tile staged transposed via VGPRs.
__global__ __launch_bounds__(256) void wmma_gemm_kernel(
    const float* __restrict__ A, const float* __restrict__ B, float* __restrict__ C,
    int M, int N, int K)
{
  __shared__ float As[128 * 32];   // A tile  [m][k]
  __shared__ float Bt[128 * 32];   // B tile transposed [n][k]

  const int tid  = threadIdx.x;
  const int lane = tid & 31;
  const int wid  = tid >> 5;
  const int wm   = wid >> 2;       // 0..1
  const int wn   = wid & 3;        // 0..3
  const int row0 = blockIdx.y * 128;
  const int col0 = blockIdx.x * 128;

  const uint32_t asLds = (uint32_t)(uintptr_t)(void*)&As[0];
  const uint64_t aBase = (uint64_t)(uintptr_t)(A + (size_t)row0 * K);

  union { v8f v; float e[8]; } acc[4][2];
#pragma unroll
  for (int i = 0; i < 4; ++i)
#pragma unroll
    for (int j = 0; j < 2; ++j)
#pragma unroll
      for (int r = 0; r < 8; ++r) acc[i][j].e[r] = 0.f;

  for (int k0 = 0; k0 < K; k0 += 32) {
    __syncthreads();
#pragma unroll
    for (int p = 0; p < 4; ++p) {
      int idx = tid + p * 256;               // 0..1023
      // A tile: async global -> LDS, 16B per lane, no VGPR round trip
      int m   = idx >> 3;                    // 128 rows, 8 float4 per row
      int k4  = (idx & 7) << 2;
      async_copy_b128(asLds + (uint32_t)((m * 32 + k4) * 4),
                      (uint32_t)((m * K + k0 + k4) * 4), aBase);
      // B tile: load + transpose into [n][k]
      int kk  = idx >> 5;                    // 32 rows, 32 float4 per row
      int n4  = (idx & 31) << 2;
      float4 bv = *(const float4*)(B + (size_t)(k0 + kk) * N + (col0 + n4));
      Bt[(n4 + 0) * 32 + kk] = bv.x;
      Bt[(n4 + 1) * 32 + kk] = bv.y;
      Bt[(n4 + 2) * 32 + kk] = bv.z;
      Bt[(n4 + 3) * 32 + kk] = bv.w;
    }
    if (k0 + 32 < K) {  // gfx1250 global_prefetch for the next B K-tile
      __builtin_prefetch(B + (size_t)(k0 + 32 + (tid >> 5)) * N + col0 + ((tid & 31) << 2), 0, 1);
    }
    wait_asynccnt0();
    __syncthreads();

    v16bf afr[4], bfr[2];
#pragma unroll
    for (int i = 0; i < 4; ++i) {
      int m = wm * 64 + i * 16 + (lane & 15);
      afr[i] = make_frag(&As[m * 32 + ((lane >> 4) << 3)]);
    }
#pragma unroll
    for (int j = 0; j < 2; ++j) {
      int n = wn * 32 + j * 16 + (lane & 15);
      bfr[j] = make_frag(&Bt[n * 32 + ((lane >> 4) << 3)]);
    }
#pragma unroll
    for (int i = 0; i < 4; ++i)
#pragma unroll
      for (int j = 0; j < 2; ++j)
        acc[i][j].v = __builtin_amdgcn_wmma_f32_16x16x32_bf16(
            false, afr[i], false, bfr[j], (short)0, acc[i][j].v, false, false);
  }

#pragma unroll
  for (int i = 0; i < 4; ++i)
#pragma unroll
    for (int j = 0; j < 2; ++j)
#pragma unroll
      for (int r = 0; r < 8; ++r) {
        int row = row0 + wm * 64 + i * 16 + r + ((lane >> 4) << 3);
        int col = col0 + wn * 32 + j * 16 + (lane & 15);
        C[(size_t)row * N + col] = acc[i][j].e[r];
      }
}

// Fused RMSNorm (over head_dim=128) + RoPE, in place. One wave32 per (b,t,head) row.
__global__ __launch_bounds__(128) void rms_rope_kernel(
    float* __restrict__ x, const float* __restrict__ scale,
    int nheads, int T, int total_rows)
{
  int row = blockIdx.x * 4 + (threadIdx.x >> 5);
  if (row >= total_rows) return;
  int lane = threadIdx.x & 31;
  int t = (row / nheads) % T;
  float* p = x + (size_t)row * 128;
  float v0 = p[lane], v1 = p[lane + 32], v2 = p[lane + 64], v3 = p[lane + 96];
  float ss = v0 * v0 + v1 * v1 + v2 * v2 + v3 * v3;
#pragma unroll
  for (int m = 1; m < 32; m <<= 1) ss += __shfl_xor(ss, m, 32);
  float rn = rsqrtf(ss * (1.f / 128.f) + 1e-6f);
  float n0 = v0 * rn * scale[lane];
  float n1 = v1 * rn * scale[lane + 32];
  float n2 = v2 * rn * scale[lane + 64];
  float n3 = v3 * rn * scale[lane + 96];
  const float kLog1e6_64 = 13.815510557964274f / 64.f;
  float tf  = (float)t;
  float th0 = tf * expf(-(float)lane * kLog1e6_64);
  float th1 = tf * expf(-(float)(lane + 32) * kLog1e6_64);
  float c0 = cosf(th0), s0 = sinf(th0), c1 = cosf(th1), s1 = sinf(th1);
  p[lane]      = n0 * c0 - n2 * s0;
  p[lane + 32] = n1 * c1 - n3 * s1;
  p[lane + 64] = n2 * c0 + n0 * s0;
  p[lane + 96] = n3 * c1 + n1 * s1;
}

// Flash attention, GQA (16 Q heads, 4 KV heads). Block = 4 waves, each wave owns
// 16 q rows; S-loop in tiles of 32 with online softmax. bf16 WMMA, f32 accumulate.
// K tile delivered by the Tensor Data Mover; V tile staged transposed + packed bf16.
__global__ __launch_bounds__(128) void flash_attn_kernel(
    const float* __restrict__ q, const float* __restrict__ k, const float* __restrict__ v,
    const unsigned char* __restrict__ mask, float* __restrict__ out, int T)
{
  __shared__ float    Ks[32 * 128];    // K tile [token][d], fp32 (TDM target)
  __shared__ uint32_t VsT[128 * 16];   // V^T tile: [d][tok/2], bf16 pairs (tok, tok+1)
  __shared__ float    Ps[4][16 * 32];  // per-wave P re-layout scratch

  const int tid  = threadIdx.x;
  const int lane = tid & 31;
  const int wv   = tid >> 5;
  const int h    = blockIdx.y;
  const int b    = blockIdx.z;
  const int hk   = h >> 2;                 // GQA: kv head = h / 4
  const int t0   = blockIdx.x * 64 + wv * 16;

  const uint32_t ksLds = (uint32_t)(uintptr_t)(void*)&Ks[0];

  // Q fragments for this wave's 16 rows (4 K-chunks of 32 over head_dim=128)
  v16bf qfr[4];
  {
    int t = t0 + (lane & 15);
    const float* qrow = q + (((size_t)(b * T + t)) * 16 + h) * 128 + ((lane >> 4) << 3);
#pragma unroll
    for (int kc = 0; kc < 4; ++kc) qfr[kc] = make_frag(qrow + kc * 32);
  }

  union { v8f v; float e[8]; } Oacc[8];
  float mrow[8], lrow[8];
#pragma unroll
  for (int j = 0; j < 8; ++j)
#pragma unroll
    for (int r = 0; r < 8; ++r) Oacc[j].e[r] = 0.f;
#pragma unroll
  for (int r = 0; r < 8; ++r) { mrow[r] = -__builtin_inff(); lrow[r] = 0.f; }

  const float scl = 0.08838834764831845f;  // 1/sqrt(128)
  float* Psw = &Ps[wv][0];

  for (int s0 = 0; s0 < T; s0 += 32) {
    __syncthreads();   // all waves done reading previous K/V tiles
    const float* kvRow0 = k + (((size_t)(b * T + s0)) * 4 + hk) * 128;
    const float* vRow0  = v + (((size_t)(b * T + s0)) * 4 + hk) * 128;
#if USE_TDM
    if (wv == 0) {
      tdm_load_2d(ksLds, (uint64_t)(uintptr_t)kvRow0, 32, 128, 512);  // 32 tok x 128 d
    }
#else
    {
      uint64_t kbase = (uint64_t)(uintptr_t)kvRow0;
#pragma unroll
      for (int p = 0; p < 8; ++p) {
        int idx = tid + p * 128;              // 0..1023
        int kk  = idx >> 5;                   // 32 tokens
        int d4  = (idx & 31) << 2;            // 128 dims
        async_copy_b128(ksLds + (uint32_t)((kk * 128 + d4) * 4),
                        (uint32_t)((kk * 512 + d4) * 4), kbase);
      }
    }
#endif
    // V tile: load fp32, convert + transpose into packed-bf16 [d][tok/2]
#pragma unroll
    for (int p = 0; p < 4; ++p) {
      int idx = tid + p * 128;                // 0..511
      int kk2 = (idx & 15) << 1;              // even token 0..30
      int d4  = ((idx >> 4) << 2);            // 0..124
      float4 a0 = *(const float4*)(vRow0 + (size_t)kk2 * 512 + d4);
      float4 a1 = *(const float4*)(vRow0 + (size_t)(kk2 + 1) * 512 + d4);
      union { __bf16 h2[2]; uint32_t u; } pk;
      pk.h2[0] = (__bf16)a0.x; pk.h2[1] = (__bf16)a1.x; VsT[(d4 + 0) * 16 + (kk2 >> 1)] = pk.u;
      pk.h2[0] = (__bf16)a0.y; pk.h2[1] = (__bf16)a1.y; VsT[(d4 + 1) * 16 + (kk2 >> 1)] = pk.u;
      pk.h2[0] = (__bf16)a0.z; pk.h2[1] = (__bf16)a1.z; VsT[(d4 + 2) * 16 + (kk2 >> 1)] = pk.u;
      pk.h2[0] = (__bf16)a0.w; pk.h2[1] = (__bf16)a1.w; VsT[(d4 + 3) * 16 + (kk2 >> 1)] = pk.u;
    }
#if USE_TDM
    if (wv == 0) __builtin_amdgcn_s_wait_tensorcnt(0);
#else
    wait_asynccnt0();
#endif
    __syncthreads();   // K + V^T tiles published to all waves

    // logits: two 16x16 C tiles covering s0..s0+31
    union { v8f v; float e[8]; } lg[2];
#pragma unroll
    for (int jt = 0; jt < 2; ++jt) {
#pragma unroll
      for (int r = 0; r < 8; ++r) lg[jt].e[r] = 0.f;
#pragma unroll
      for (int kc = 0; kc < 4; ++kc) {
        v16bf bfr = make_frag(&Ks[(jt * 16 + (lane & 15)) * 128 + kc * 32 + ((lane >> 4) << 3)]);
        lg[jt].v = __builtin_amdgcn_wmma_f32_16x16x32_bf16(
            false, qfr[kc], false, bfr, (short)0, lg[jt].v, false, false);
      }
#pragma unroll
      for (int r = 0; r < 8; ++r) {
        int t = t0 + r + ((lane >> 4) << 3);
        int s = s0 + jt * 16 + (lane & 15);
        float x = lg[jt].e[r] * scl;
        if (!mask[((size_t)(b * T + t)) * T + s]) x = -1e30f;
        lg[jt].e[r] = x;
      }
    }

    // online softmax (rows live in vgpr r across a 16-lane half)
    float alpha[8];
#pragma unroll
    for (int r = 0; r < 8; ++r) {
      float tm = fmaxf(lg[0].e[r], lg[1].e[r]);
#pragma unroll
      for (int m = 1; m < 16; m <<= 1) tm = fmaxf(tm, __shfl_xor(tm, m, 32));
      float nm = fmaxf(mrow[r], tm);
      float a  = (mrow[r] == nm) ? 1.f : expf(mrow[r] - nm);
      float p0 = expf(lg[0].e[r] - nm);
      float p1 = expf(lg[1].e[r] - nm);
      float rs = p0 + p1;
#pragma unroll
      for (int m = 1; m < 16; m <<= 1) rs += __shfl_xor(rs, m, 32);
      lrow[r] = lrow[r] * a + rs;
      mrow[r] = nm;
      alpha[r] = a;
      int mrw = r + ((lane >> 4) << 3);
      Psw[mrw * 32 + (lane & 15)]      = p0;   // C layout -> [row][s] scratch
      Psw[mrw * 32 + 16 + (lane & 15)] = p1;
    }
#pragma unroll
    for (int j = 0; j < 8; ++j)
#pragma unroll
      for (int r = 0; r < 8; ++r) Oacc[j].e[r] *= alpha[r];

    // P as A-fragment (wave-private LDS region; DS ops in-order within a wave)
    v16bf pfr = make_frag(&Psw[(lane & 15) * 32 + ((lane >> 4) << 3)]);
    // V fragments: pure b128 loads from packed-bf16 V^T, no converts
#pragma unroll
    for (int j = 0; j < 8; ++j) {
      union { v16bf v; uint4 q2[2]; } vf;
      const uint32_t* src = &VsT[(j * 16 + (lane & 15)) * 16 + ((lane >> 4) << 2)];
      vf.q2[0] = *(const uint4*)(src);       // K 0..7   (bf16 tokens, interleave low)
      vf.q2[1] = *(const uint4*)(src + 8);   // K 16..23 (interleave high)
      Oacc[j].v = __builtin_amdgcn_wmma_f32_16x16x32_bf16(
          false, pfr, false, vf.v, (short)0, Oacc[j].v, false, false);
    }
  }

#pragma unroll
  for (int r = 0; r < 8; ++r) {
    float inv = 1.f / lrow[r];
    int t = t0 + r + ((lane >> 4) << 3);
#pragma unroll
    for (int j = 0; j < 8; ++j)
      out[(((size_t)(b * T + t)) * 16 + h) * 128 + j * 16 + (lane & 15)] = Oacc[j].e[r] * inv;
  }
}

extern "C" void kernel_launch(void* const* d_in, const int* in_sizes, int n_in,
                              void* d_out, int out_size, void* d_ws, size_t ws_size,
                              hipStream_t stream) {
  (void)in_sizes; (void)n_in; (void)out_size; (void)ws_size;
  const float*         x    = (const float*)d_in[0];
  const unsigned char* mask = (const unsigned char*)d_in[1];
  const float*         Wq   = (const float*)d_in[2];
  const float*         Wk   = (const float*)d_in[3];
  const float*         Wv   = (const float*)d_in[4];
  const float*         qs   = (const float*)d_in[5];
  const float*         ks   = (const float*)d_in[6];
  const float*         Wo   = (const float*)d_in[7];
  float* out = (float*)d_out;

  const int B = 2, T = 2048, H = 2048, NH = 16, NKV = 4, HD = 128;
  const int M = B * T;  // 4096 rows

  float* ws = (float*)d_ws;
  float* qb = ws;                                  // [M, 16*128]
  float* kb = qb + (size_t)M * NH * HD;            // [M, 4*128]
  float* vb = kb + (size_t)M * NKV * HD;           // [M, 4*128]
  float* ab = vb + (size_t)M * NKV * HD;           // [M, 16*128]

  dim3 blk(256);
  // QKV projections
  wmma_gemm_kernel<<<dim3(NH * HD / 128,  M / 128), blk, 0, stream>>>(x, Wq, qb, M, NH * HD, H);
  wmma_gemm_kernel<<<dim3(NKV * HD / 128, M / 128), blk, 0, stream>>>(x, Wk, kb, M, NKV * HD, H);
  wmma_gemm_kernel<<<dim3(NKV * HD / 128, M / 128), blk, 0, stream>>>(x, Wv, vb, M, NKV * HD, H);
  // RMSNorm + RoPE (in place)
  rms_rope_kernel<<<dim3(M * NH / 4),  dim3(128), 0, stream>>>(qb, qs, NH,  T, M * NH);
  rms_rope_kernel<<<dim3(M * NKV / 4), dim3(128), 0, stream>>>(kb, ks, NKV, T, M * NKV);
  // Flash attention
  flash_attn_kernel<<<dim3(T / 64, NH, B), dim3(128), 0, stream>>>(qb, kb, vb, mask, ab, T);
  // Output projection
  wmma_gemm_kernel<<<dim3(H / 128, M / 128), blk, 0, stream>>>(ab, Wo, out, M, H, H);
}
